// ContrastiveAdditiveIndependentScorer_85263690760579
// MI455X (gfx1250) — matrix operand order repaired and checked by reading, not compiled, not run on hardware
//
#include <hip/hip_runtime.h>
#include <hip/hip_bf16.h>

#define BATCH 8192
#define DDIM 64
#define TDIM 16
#define HDIM 128
#define FDIM 16

#define BM 128         // batch rows per workgroup tile
#define NTHREADS 256   // 8 waves (wave32)
#define AST 136        // activation LDS stride (bf16 elems), padded vs 128
#define WST 136        // weight LDS stride (bf16 elems), padded vs 128

typedef __bf16 v16bf __attribute__((ext_vector_type(16)));
typedef __bf16 v8bf  __attribute__((ext_vector_type(8)));
typedef float  v8f   __attribute__((ext_vector_type(8)));

// silu via fast hardware reciprocal: v_exp_f32 + v_rcp_f32 (no IEEE div sequence)
__device__ __forceinline__ float silu_f(float v) {
  return v * __builtin_amdgcn_rcpf(1.0f + __expf(-v));
}

__device__ __forceinline__ v16bf cat16(v8bf lo, v8bf hi) {
  return __builtin_shufflevector(lo, hi, 0,1,2,3,4,5,6,7,8,9,10,11,12,13,14,15);
}

// One 16-row strip (rows m0..m0+15) times WT (n-major in LDS), NTILES 16-col tiles,
// K = KTILES*32.  A layout: lane ln=row, halves g=0/1 carry K {g*8..g*8+7} and
// {16+g*8..23+g*8} per 32-chunk.  B layout: lane col = ln, same K split as A.
template<int KTILES, int NTILES>
__device__ __forceinline__ void gemm_inner(const __bf16* act, const __bf16* wt,
                                           int m0, int ln, int g, v8f acc[NTILES]) {
#pragma unroll
  for (int nt = 0; nt < NTILES; ++nt) { v8f z = {}; acc[nt] = z; }
  const __bf16* arow = act + (m0 + ln) * AST;
#pragma unroll
  for (int kt = 0; kt < KTILES; ++kt) {
    const int kc = kt * 32;
    v8bf alo = *(const v8bf*)(arow + kc + g * 8);
    v8bf ahi = *(const v8bf*)(arow + kc + 16 + g * 8);
    v16bf afrag = cat16(alo, ahi);
#pragma unroll
    for (int nt = 0; nt < NTILES; ++nt) {
      const __bf16* bp = wt + (nt * 16 + ln) * WST + kc + g * 8;
      v16bf bfrag = cat16(*(const v8bf*)bp, *(const v8bf*)(bp + 16));
      acc[nt] = __builtin_amdgcn_wmma_f32_16x16x32_bf16(
          false, afrag, false, bfrag, (short)0, acc[nt], false, false);
    }
  }
}

// Full BM x 128 layer: GEMM + bias (+silu) written back in place into act (bf16).
// C layout: VGPR r holds row m0+g*8+r, lane column n = nt*16+ln.
template<int KTILES, bool SILU>
__device__ __forceinline__ void gemm_layer_inplace(__bf16* act, const __bf16* wt,
                                                   const float* bias,
                                                   int m0, int ln, int g) {
  v8f acc[8];
  gemm_inner<KTILES, 8>(act, wt, m0, ln, g, acc);
  __syncthreads();  // all waves done reading act before in-place overwrite
#pragma unroll
  for (int nt = 0; nt < 8; ++nt) {
    const int n = nt * 16 + ln;
    const float bv = bias[n];
#pragma unroll
    for (int r = 0; r < 8; ++r) {
      float v = acc[nt][r] + bv;
      if (SILU) v = silu_f(v);
      act[(m0 + g * 8 + r) * AST + n] = (__bf16)v;
    }
  }
  __syncthreads();
}

// ---------------- theta MLP: at[B,16] -------------------------------------
__global__ __launch_bounds__(NTHREADS)
void theta_mlp_kernel(const float* __restrict__ theta,
                      const float* __restrict__ A1, const float* __restrict__ ab1,
                      const float* __restrict__ A2, const float* __restrict__ ab2,
                      const float* __restrict__ A3, const float* __restrict__ ab3,
                      const float* __restrict__ Ao, const float* __restrict__ abo,
                      float* __restrict__ at) {
  __shared__ __bf16 s_act[BM * AST];
  __shared__ __bf16 s_wt[HDIM * WST];
  __shared__ float  s_bias[HDIM];

  const int tid = threadIdx.x;
  const int lane = tid & 31, wave = tid >> 5;
  const int g = lane >> 4, ln = lane & 15;
  const int m0 = wave * 16;
  const int b0 = blockIdx.x * BM;

  // stage theta tile (BM x 32, K zero-padded 16->32)
  for (int i = tid; i < BM * 32; i += NTHREADS) {
    int row = i >> 5, col = i & 31;
    float v = (col < TDIM) ? theta[(size_t)(b0 + row) * TDIM + col] : 0.0f;
    s_act[row * AST + col] = (__bf16)v;
  }
  // stage A1^T (128 x 32, K zero-padded)
  for (int i = tid; i < HDIM * 32; i += NTHREADS) {
    int n = i >> 5, k = i & 31;
    s_wt[n * WST + k] = (k < TDIM) ? (__bf16)A1[k * HDIM + n] : (__bf16)0.0f;
  }
  if (tid < HDIM) s_bias[tid] = ab1[tid];
  __syncthreads();
  gemm_layer_inplace<1, true>(s_act, s_wt, s_bias, m0, ln, g);

  for (int i = tid; i < HDIM * HDIM; i += NTHREADS) {
    int h = i >> 7, k = i & 127;
    s_wt[k * WST + h] = (__bf16)A2[i];
  }
  if (tid < HDIM) s_bias[tid] = ab2[tid];
  __syncthreads();
  gemm_layer_inplace<4, true>(s_act, s_wt, s_bias, m0, ln, g);

  for (int i = tid; i < HDIM * HDIM; i += NTHREADS) {
    int h = i >> 7, k = i & 127;
    s_wt[k * WST + h] = (__bf16)A3[i];
  }
  if (tid < HDIM) s_bias[tid] = ab3[tid];
  __syncthreads();
  gemm_layer_inplace<4, true>(s_act, s_wt, s_bias, m0, ln, g);

  // Ao^T (16 x 128)
  for (int i = tid; i < HDIM * FDIM; i += NTHREADS) {
    int h = i >> 4, f = i & 15;
    s_wt[f * WST + h] = (__bf16)Ao[i];
  }
  if (tid < FDIM) s_bias[tid] = abo[tid];
  __syncthreads();

  v8f acc[1];
  gemm_inner<4, 1>(s_act, s_wt, m0, ln, g, acc);
  const float bv = s_bias[ln];
#pragma unroll
  for (int r = 0; r < 8; ++r)
    at[(size_t)(b0 + m0 + g * 8 + r) * FDIM + ln] = acc[0][r] + bv;
}

// ---------------- per-dim MLP + fused dot ---------------------------------
__global__ __launch_bounds__(NTHREADS)
void scorer_main_kernel(const float* __restrict__ x,
                        const float* __restrict__ W1, const float* __restrict__ b1,
                        const float* __restrict__ W2, const float* __restrict__ b2,
                        const float* __restrict__ W3, const float* __restrict__ b3,
                        const float* __restrict__ Wo, const float* __restrict__ bo,
                        const float* __restrict__ at, float* __restrict__ partial) {
  __shared__ __bf16 s_act[BM * AST];
  __shared__ __bf16 s_wt[HDIM * WST];
  __shared__ float  s_bias[HDIM];
  __shared__ float  s_x[BM];
  __shared__ float  s_w1[HDIM];
  __shared__ float  s_b1[HDIM];

  const int tid = threadIdx.x;
  const int lane = tid & 31, wave = tid >> 5;
  const int g = lane >> 4, ln = lane & 15;
  const int m0 = wave * 16;
  const int b0 = blockIdx.x * BM;
  const int d  = blockIdx.y;

  const float* W2d = W2 + (size_t)d * HDIM * HDIM;
  const float* W3d = W3 + (size_t)d * HDIM * HDIM;
  const float* Wod = Wo + (size_t)d * HDIM * FDIM;

  // stage W2^T -> LDS bf16 (n-major so B-fragments are contiguous)
  for (int i = tid; i < HDIM * HDIM; i += NTHREADS) {
    int h = i >> 7, k = i & 127;
    s_wt[k * WST + h] = (__bf16)W2d[i];
  }
  if (tid < HDIM) {
    s_bias[tid] = b2[d * HDIM + tid];
    s_w1[tid]   = W1[d * HDIM + tid];
    s_b1[tid]   = b1[d * HDIM + tid];
  }
  if (tid < BM) s_x[tid] = x[(size_t)(b0 + tid) * DDIM + d];
  // pull next layer's weights toward L2 while we compute (global_prefetch_b8)
  __builtin_prefetch(W3d + tid * 64, 0, 1);
  __builtin_prefetch(W3d + tid * 64 + 32, 0, 1);
  __syncthreads();

  // layer 1: h = silu(x_d * W1[d] + b1[d])  (rank-1, VALU)
  for (int i = tid; i < BM * HDIM; i += NTHREADS) {
    int row = i >> 7, col = i & 127;
    s_act[row * AST + col] = (__bf16)silu_f(s_x[row] * s_w1[col] + s_b1[col]);
  }
  __syncthreads();

  // layer 2
  gemm_layer_inplace<4, true>(s_act, s_wt, s_bias, m0, ln, g);

  // stage W3^T, b3
  for (int i = tid; i < HDIM * HDIM; i += NTHREADS) {
    int h = i >> 7, k = i & 127;
    s_wt[k * WST + h] = (__bf16)W3d[i];
  }
  if (tid < HDIM) s_bias[tid] = b3[d * HDIM + tid];
  __builtin_prefetch(Wod + tid * 8, 0, 1);
  __syncthreads();

  // layer 3
  gemm_layer_inplace<4, true>(s_act, s_wt, s_bias, m0, ln, g);

  // stage Wo^T (16 x 128), bo
  for (int i = tid; i < HDIM * FDIM; i += NTHREADS) {
    int h = i >> 4, f = i & 15;
    s_wt[f * WST + h] = (__bf16)Wod[i];
  }
  if (tid < FDIM) s_bias[tid] = bo[d * FDIM + tid];
  __syncthreads();

  // output layer (N=16) fused with <hx, at> dot; reduce across 16 lanes (f)
  v8f acc[1];
  gemm_inner<4, 1>(s_act, s_wt, m0, ln, g, acc);
  const float bv = s_bias[ln];
#pragma unroll
  for (int r = 0; r < 8; ++r) {
    const int row = b0 + m0 + g * 8 + r;
    float v = (acc[0][r] + bv) * at[(size_t)row * FDIM + ln];
    v += __shfl_xor(v, 8);
    v += __shfl_xor(v, 4);
    v += __shfl_xor(v, 2);
    v += __shfl_xor(v, 1);
    if (ln == 0) partial[(size_t)d * BATCH + row] = v;
  }
}

// ---------------- deterministic reduction over d --------------------------
__global__ void reduce_kernel(const float* __restrict__ partial,
                              float* __restrict__ out) {
  int b = blockIdx.x * blockDim.x + threadIdx.x;
  if (b >= BATCH) return;
  float s = 0.0f;
#pragma unroll 8
  for (int d = 0; d < DDIM; ++d) s += partial[(size_t)d * BATCH + b];
  out[b] = s * (1.0f / (float)DDIM);
}

extern "C" void kernel_launch(void* const* d_in, const int* in_sizes, int n_in,
                              void* d_out, int out_size, void* d_ws, size_t ws_size,
                              hipStream_t stream) {
  const float* x   = (const float*)d_in[0];
  const float* th  = (const float*)d_in[1];
  const float* W1  = (const float*)d_in[2];
  const float* b1  = (const float*)d_in[3];
  const float* W2  = (const float*)d_in[4];
  const float* b2  = (const float*)d_in[5];
  const float* W3  = (const float*)d_in[6];
  const float* b3  = (const float*)d_in[7];
  const float* Wo  = (const float*)d_in[8];
  const float* bo  = (const float*)d_in[9];
  const float* A1  = (const float*)d_in[10];
  const float* ab1 = (const float*)d_in[11];
  const float* A2  = (const float*)d_in[12];
  const float* ab2 = (const float*)d_in[13];
  const float* A3  = (const float*)d_in[14];
  const float* ab3 = (const float*)d_in[15];
  const float* Ao  = (const float*)d_in[16];
  const float* abo = (const float*)d_in[17];
  float* out = (float*)d_out;

  float* at_ws   = (float*)d_ws;                   // B*F floats
  float* partial = at_ws + (size_t)BATCH * FDIM;   // D*B floats

  theta_mlp_kernel<<<BATCH / BM, NTHREADS, 0, stream>>>(
      th, A1, ab1, A2, ab2, A3, ab3, Ao, abo, at_ws);

  dim3 grid(BATCH / BM, DDIM);
  scorer_main_kernel<<<grid, NTHREADS, 0, stream>>>(
      x, W1, b1, W2, b2, W3, b3, Wo, bo, at_ws, partial);

  reduce_kernel<<<(BATCH + 255) / 256, 256, 0, stream>>>(partial, out);
}
